// DeformNet_46815143526475
// MI455X (gfx1250) — matrix operand order repaired and checked by reading, not compiled, not run on hardware
//
#include <hip/hip_runtime.h>
#include <hip/hip_bf16.h>

// DeformNet forward for MI455X (gfx1250, wave32, WMMA).
// Exploits the exact rank-1 structure of the attention keys/values (k,v are
// affine in loc_scores), collapsing the O(N^2) attention into per-(head,n)
// softmax-weighted means of loc_scores. Big GEMMs run on
// v_wmma_f32_16x16x32_bf16 with f32 accumulation (bf16 packed into LDS via
// v_cvt_pk_bf16_f32, staged with b32/b64 ds stores); the normal-equations
// build for the spectral solve uses exact-fp32 v_wmma_f32_16x16x4_f32.
// loc_scores staging uses GLOBAL_LOAD_ASYNC_TO_LDS_B128 when available.

constexpr int N_  = 6144;
constexpr int M_  = 6144;
constexpr int C_  = 256;
constexpr int C2_ = 512;
constexpr int K_  = 128;

typedef __attribute__((ext_vector_type(16))) __bf16 v16bf;
typedef __attribute__((ext_vector_type(8)))  float  v8f;
typedef __attribute__((ext_vector_type(2)))  float  v2f;
typedef __attribute__((ext_vector_type(2)))  __bf16 bf2;
typedef int v4i_ __attribute__((vector_size(16)));

union AFrag { v16bf v; bf2 p[8]; };

static __device__ __forceinline__ float wave_red(float v) {
#pragma unroll
  for (int o = 16; o; o >>= 1) v += __shfl_xor(v, o, 32);
  return v;
}

// ---------------------------------------------------------------------------
// F[n][c] = sum_m Pxy[n][m]*feat_y[m][c] - feat_x[n][c]   (bf16 WMMA, f32 acc)
// grid(4,96) block(256): 64x64 tile; wave (wn,wc) does rows wn*32..+31 x 16 cols
// Also produces per-block partial sums of F^2 (deterministic sigma^2).
// ---------------------------------------------------------------------------
__global__ __launch_bounds__(256) void k_gemmF(
    const float* __restrict__ Pxy, const float* __restrict__ feat_y,
    const float* __restrict__ feat_x, float* __restrict__ F, float* __restrict__ part)
{
  const int c0 = blockIdx.x * 64;
  const int n0 = blockIdx.y * 64;
  __shared__ bf2 As[64][20];   // [n_local][k/2]  pitch 20 dwords (bank-spread)
  __shared__ bf2 Bs[64][22];   // [c_local][k/2]  pitch 22 dwords
  __shared__ float wred[8];
  const int tid = threadIdx.x, lane = tid & 31, wave = tid >> 5, g = lane >> 4;
  const int wn = wave & 1, wc = wave >> 1;
  v8f acc0 = {0.f,0.f,0.f,0.f,0.f,0.f,0.f,0.f};
  v8f acc1 = {0.f,0.f,0.f,0.f,0.f,0.f,0.f,0.f};

  for (int k0 = 0; k0 < M_; k0 += 32) {
    __syncthreads();
    // A tile: 64 rows x 32 k of Pxy; float4 loads, packed bf16x2 LDS stores
#pragma unroll
    for (int e = 0; e < 2; ++e) {
      int q = tid + e * 256;                  // 512 quads
      int r = q >> 3, k4 = (q & 7) << 2;
      const float* srcp = Pxy + (size_t)(n0 + r) * M_ + k0 + k4;
      const float4 v = *(const float4*)srcp;
      bf2 p0, p1;
      p0.x = (__bf16)v.x; p0.y = (__bf16)v.y;
      p1.x = (__bf16)v.z; p1.y = (__bf16)v.w;
      As[r][(k4 >> 1) + 0] = p0;
      As[r][(k4 >> 1) + 1] = p1;
      if (e == 0) __builtin_prefetch(srcp + 256, 0, 0);   // ~8 K-steps ahead
    }
    // B tile: 32 k x 64 c of feat_y, transposed; pairs packed along k
#pragma unroll
    for (int e = 0; e < 4; ++e) {
      int p = tid + e * 256;                  // 1024 pairs
      int j = p & 63, r2 = p >> 6;            // r2 in 0..15
      const float* src = feat_y + (size_t)(k0 + 2 * r2) * C_ + c0 + j;
      bf2 t; t.x = (__bf16)src[0]; t.y = (__bf16)src[C_];
      Bs[j][r2] = t;
    }
    __syncthreads();
    AFrag a0, a1, b;
    const int ar = wn * 32 + (lane & 15);
    const int bc = wc * 16 + (lane & 15);
#pragma unroll
    for (int t = 0; t < 4; ++t) {
      a0.p[t]     = As[ar][4*g + t];
      a0.p[4 + t] = As[ar][8 + 4*g + t];
      a1.p[t]     = As[ar + 16][4*g + t];
      a1.p[4 + t] = As[ar + 16][8 + 4*g + t];
    }
#pragma unroll
    for (int t = 0; t < 4; ++t) { b.p[t] = Bs[bc][4*g + t]; b.p[4 + t] = Bs[bc][8 + 4*g + t]; }
    acc0 = __builtin_amdgcn_wmma_f32_16x16x32_bf16(false, a0.v, false, b.v, (short)0, acc0, false, false);
    acc1 = __builtin_amdgcn_wmma_f32_16x16x32_bf16(false, a1.v, false, b.v, (short)0, acc1, false, false);
  }
  const int c = c0 + wc * 16 + (lane & 15);
  float ss = 0.f;
#pragma unroll
  for (int r = 0; r < 8; ++r) {
    int n = n0 + wn * 32 + r + 8 * g;
    float v0 = acc0[r] - feat_x[(size_t)n * C_ + c];
    F[(size_t)n * C_ + c] = v0; ss += v0 * v0;
    int n2 = n + 16;
    float v1 = acc1[r] - feat_x[(size_t)n2 * C_ + c];
    F[(size_t)n2 * C_ + c] = v1; ss += v1 * v1;
  }
  ss = wave_red(ss);
  if (lane == 0) wred[wave] = ss;
  __syncthreads();
  if (tid == 0) {
    float s = 0.f;
#pragma unroll
    for (int i = 0; i < 8; ++i) s += wred[i];
    part[blockIdx.y * 4 + blockIdx.x] = s;   // 384 partials
  }
}

__global__ __launch_bounds__(128) void k_sumfin(const float* __restrict__ part, float* __restrict__ sumF2) {
  float s = 0.f;
  for (int i = threadIdx.x; i < 384; i += 128) s += part[i];
  __shared__ float red[128];
  red[threadIdx.x] = s; __syncthreads();
  for (int o = 64; o; o >>= 1) { if ((int)threadIdx.x < o) red[threadIdx.x] += red[threadIdx.x + o]; __syncthreads(); }
  if (threadIdx.x == 0) sumF2[0] = red[0];
}

// vert_F[n][j] = sum_m Pxy[n][m]*vert_y[m][j] - vert_x[n][j]
__global__ __launch_bounds__(256) void k_vertF(
    const float* __restrict__ Pxy, const float* __restrict__ vert_y,
    const float* __restrict__ vert_x, float* __restrict__ vertF)
{
  const int n = blockIdx.x, tid = threadIdx.x;
  float s0 = 0.f, s1 = 0.f, s2 = 0.f;
  for (int m = tid; m < M_; m += 256) {
    float p = Pxy[(size_t)n * M_ + m];
    s0 += p * vert_y[m*3+0]; s1 += p * vert_y[m*3+1]; s2 += p * vert_y[m*3+2];
  }
  __shared__ float red[3][256];
  red[0][tid] = s0; red[1][tid] = s1; red[2][tid] = s2; __syncthreads();
  for (int o = 128; o; o >>= 1) {
    if (tid < o) { red[0][tid]+=red[0][tid+o]; red[1][tid]+=red[1][tid+o]; red[2][tid]+=red[2][tid+o]; }
    __syncthreads();
  }
  if (tid < 3) vertF[n*3 + tid] = red[tid][0] - vert_x[n*3 + tid];
}

// ---------------------------------------------------------------------------
// Precompute collapsed-attention constants (see header comment).
// ---------------------------------------------------------------------------
__global__ __launch_bounds__(512) void k_pre(
    const float* __restrict__ wq, const float* __restrict__ bq,
    const float* __restrict__ wk, const float* __restrict__ wv, const float* __restrict__ bv,
    const float* __restrict__ wmh, const float* __restrict__ bmh,
    const float* __restrict__ wc1, const float* __restrict__ bc1,
    const float* __restrict__ w_up, const float* __restrict__ b_up,
    float* __restrict__ u, float* __restrict__ tvec,
    float* __restrict__ G1, float* __restrict__ e1)
{
  __shared__ float ak[C_], av[C_], cv[C_], gm[C_ * 4], dv[C_];
  const int tid = threadIdx.x;
  if (tid < C_) {
    float sk = 0.f, sv = 0.f, sc = 0.f;
    for (int c = 0; c < C_; ++c) {
      float wu = w_up[c], bu = b_up[c];
      sk += wk[tid*C_ + c] * wu;
      sv += wv[tid*C_ + c] * wu;
      sc += wv[tid*C_ + c] * bu;
    }
    ak[tid] = sk; av[tid] = sv; cv[tid] = sc + bv[tid];
  }
  __syncthreads();
  for (int idx = tid; idx < 4 * C_; idx += 512) {
    int h = idx >> 8, c = idx & (C_ - 1);
    float s = 0.f;
    for (int j = 0; j < 64; ++j) s += ak[h*64 + j] * wq[(h*64 + j)*C_ + c];
    u[idx] = s * 0.125f;
  }
  if (tid < 4) {
    float s = 0.f;
    for (int j = 0; j < 64; ++j) s += ak[tid*64 + j] * bq[tid*64 + j];
    tvec[tid] = s * 0.125f;
  }
  if (tid < C_) {
    float d = 0.f;
    for (int h = 0; h < 4; ++h) {
      float s = 0.f;
      for (int j = 0; j < 64; ++j) s += wmh[tid*C_ + h*64 + j] * av[h*64 + j];
      gm[tid*4 + h] = s;
    }
    for (int o = 0; o < C_; ++o) d += wmh[tid*C_ + o] * cv[o];
    dv[tid] = d + bmh[tid];
  }
  __syncthreads();
  if (tid < C2_) {
    float e = bc1[tid], g0 = 0.f, g1 = 0.f, g2 = 0.f, g3 = 0.f;
    for (int c = 0; c < C_; ++c) {
      float w = wc1[(size_t)tid * C2_ + C_ + c];
      e  += w * dv[c];
      g0 += w * gm[c*4+0]; g1 += w * gm[c*4+1]; g2 += w * gm[c*4+2]; g3 += w * gm[c*4+3];
    }
    G1[tid*4+0] = g0; G1[tid*4+1] = g1; G1[tid*4+2] = g2; G1[tid*4+3] = g3;
    e1[tid] = e;
  }
}

// alpha[h][n] = u_h . F[n][:] + t_h
__global__ __launch_bounds__(256) void k_alpha(
    const float* __restrict__ F, const float* __restrict__ u,
    const float* __restrict__ tvec, float* __restrict__ alpha)
{
  const int n = blockIdx.x, tid = threadIdx.x;
  __shared__ float Fr[C_], red[C_];
  Fr[tid] = F[(size_t)n * C_ + tid];
  __syncthreads();
  for (int h = 0; h < 4; ++h) {
    red[tid] = u[h*C_ + tid] * Fr[tid];
    __syncthreads();
    for (int o = 128; o; o >>= 1) { if (tid < o) red[tid] += red[tid + o]; __syncthreads(); }
    if (tid == 0) alpha[h * N_ + n] = red[0] + tvec[h];
    __syncthreads();
  }
}

// Es[idx] = sum_m exp(alpha*s_m)*s_m / sum_m exp(alpha*s_m), one wave per (h,n)
__global__ __launch_bounds__(256) void k_es(
    const float* __restrict__ alpha, const float* __restrict__ s, float* __restrict__ Es)
{
  __shared__ float ss[M_];
  const int tid = threadIdx.x;
#if __has_builtin(__builtin_amdgcn_global_load_async_to_lds_b128)
  {
    for (int ch = tid * 4; ch < M_; ch += 256 * 4)
      __builtin_amdgcn_global_load_async_to_lds_b128(
          (v4i_*)(s + ch), (v4i_*)(ss + ch), 0, 0);
#if __has_builtin(__builtin_amdgcn_s_wait_asynccnt)
    __builtin_amdgcn_s_wait_asynccnt(0);
#else
    asm volatile("s_wait_asynccnt 0x0" ::: "memory");
#endif
  }
#else
  for (int m = tid; m < M_; m += 256) ss[m] = s[m];
#endif
  __syncthreads();
  const int lane = tid & 31, wave = tid >> 5;
  const int idx = blockIdx.x * 8 + wave;  // flat over 4*N_
  const float a = alpha[idx];
  float den = 0.f, num = 0.f;
  for (int m = lane; m < M_; m += 32) {
    float sv = ss[m];
    float e = __expf(a * sv);
    den += e; num += e * sv;
  }
  den = wave_red(den); num = wave_red(num);
  if (lane == 0) Es[idx] = num / den;
}

// ---------------------------------------------------------------------------
// h1[o][n] = wc1[:, :C]@F + G1.Es + e1       (bf16 WMMA)
// grid(4,384) block(256): o0=bx*128 (8 waves x 16 rows), n0=by*16
// ---------------------------------------------------------------------------
__global__ __launch_bounds__(256) void k_h1(
    const float* __restrict__ wc1, const float* __restrict__ F,
    const float* __restrict__ Es, const float* __restrict__ G1,
    const float* __restrict__ e1, float* __restrict__ h1)
{
  const int o0 = blockIdx.x * 128;
  const int n0 = blockIdx.y * 16;
  __shared__ bf2 As[128][20];  // [o_local][k/2]
  __shared__ bf2 Bs[16][22];   // [n_local][k/2]
  const int tid = threadIdx.x, lane = tid & 31, wave = tid >> 5, g = lane >> 4;
  v8f acc = {0.f,0.f,0.f,0.f,0.f,0.f,0.f,0.f};

  for (int k0 = 0; k0 < C_; k0 += 32) {
    __syncthreads();
#pragma unroll
    for (int e = 0; e < 4; ++e) {
      int q = tid + e * 256;                 // 1024 quads
      int r = q >> 3, k4 = (q & 7) << 2;
      const float4 v = *(const float4*)(wc1 + (size_t)(o0 + r) * C2_ + k0 + k4);
      bf2 p0, p1;
      p0.x = (__bf16)v.x; p0.y = (__bf16)v.y;
      p1.x = (__bf16)v.z; p1.y = (__bf16)v.w;
      As[r][(k4 >> 1) + 0] = p0;
      As[r][(k4 >> 1) + 1] = p1;
    }
    if (tid < 128) {                         // 128 quads for B
      int r = tid >> 3, k4 = (tid & 7) << 2;
      const float4 v = *(const float4*)(F + (size_t)(n0 + r) * C_ + k0 + k4);
      bf2 p0, p1;
      p0.x = (__bf16)v.x; p0.y = (__bf16)v.y;
      p1.x = (__bf16)v.z; p1.y = (__bf16)v.w;
      Bs[r][(k4 >> 1) + 0] = p0;
      Bs[r][(k4 >> 1) + 1] = p1;
    }
    __syncthreads();
    AFrag a, b;
    const int ar = wave * 16 + (lane & 15);
    const int bc = lane & 15;
#pragma unroll
    for (int t = 0; t < 4; ++t) { a.p[t] = As[ar][4*g + t]; a.p[4 + t] = As[ar][8 + 4*g + t]; }
#pragma unroll
    for (int t = 0; t < 4; ++t) { b.p[t] = Bs[bc][4*g + t]; b.p[4 + t] = Bs[bc][8 + 4*g + t]; }
    acc = __builtin_amdgcn_wmma_f32_16x16x32_bf16(false, a.v, false, b.v, (short)0, acc, false, false);
  }
  const int n = n0 + (lane & 15);
  const float e0 = Es[0*N_+n], es1 = Es[1*N_+n], es2 = Es[2*N_+n], es3 = Es[3*N_+n];
#pragma unroll
  for (int r = 0; r < 8; ++r) {
    int o = o0 + wave * 16 + r + 8 * g;
    float v = acc[r] + G1[o*4+0]*e0 + G1[o*4+1]*es1 + G1[o*4+2]*es2 + G1[o*4+3]*es3 + e1[o];
    h1[(size_t)o * N_ + n] = v;
  }
}

// per-channel batch stats (training-mode BN)
__global__ __launch_bounds__(256) void k_stats(
    const float* __restrict__ X, float* __restrict__ mean, float* __restrict__ istd)
{
  const int ch = blockIdx.x, tid = threadIdx.x;
  const float* row = X + (size_t)ch * N_;
  float s = 0.f, s2 = 0.f;
  for (int i = tid; i < N_; i += 256) { float v = row[i]; s += v; s2 += v * v; }
  __shared__ float rs[256], rq[256];
  rs[tid] = s; rq[tid] = s2; __syncthreads();
  for (int o = 128; o; o >>= 1) { if (tid < o) { rs[tid]+=rs[tid+o]; rq[tid]+=rq[tid+o]; } __syncthreads(); }
  if (tid == 0) {
    float m = rs[0] / (float)N_;
    float var = rq[0] / (float)N_ - m * m;
    mean[ch] = m; istd[ch] = rsqrtf(var + 1e-5f);
  }
}

// ---------------------------------------------------------------------------
// fp[c][n] = F[n][c] + wc2 @ relu(bn(h1)) + bc2[c]   (bf16 WMMA, BN fused)
// grid(2,384) block(256)
// ---------------------------------------------------------------------------
__global__ __launch_bounds__(256) void k_fp(
    const float* __restrict__ wc2, const float* __restrict__ h1,
    const float* __restrict__ m1, const float* __restrict__ is1,
    const float* __restrict__ g1, const float* __restrict__ b1,
    const float* __restrict__ F, const float* __restrict__ bc2, float* __restrict__ fp)
{
  const int c0 = blockIdx.x * 128;
  const int n0 = blockIdx.y * 16;
  __shared__ bf2 As[128][20];  // [c_local][k/2]
  __shared__ bf2 Bs[16][22];   // [n_local][k/2]  (transposed, BN+relu applied)
  const int tid = threadIdx.x, lane = tid & 31, wave = tid >> 5, g = lane >> 4;
  v8f acc = {0.f,0.f,0.f,0.f,0.f,0.f,0.f,0.f};

  for (int k0 = 0; k0 < C2_; k0 += 32) {
    __syncthreads();
#pragma unroll
    for (int e = 0; e < 4; ++e) {
      int q = tid + e * 256;
      int r = q >> 3, k4 = (q & 7) << 2;
      const float4 v = *(const float4*)(wc2 + (size_t)(c0 + r) * C2_ + k0 + k4);
      bf2 p0, p1;
      p0.x = (__bf16)v.x; p0.y = (__bf16)v.y;
      p1.x = (__bf16)v.z; p1.y = (__bf16)v.w;
      As[r][(k4 >> 1) + 0] = p0;
      As[r][(k4 >> 1) + 1] = p1;
    }
    {                                        // 256 pairs for B (transpose + BN)
      int r2 = tid >> 4, j = tid & 15;       // r2: k-pair 0..15, j: n_local
      int kg = k0 + 2 * r2;
      float x0 = h1[(size_t)kg * N_ + n0 + j];
      float x1 = h1[(size_t)(kg + 1) * N_ + n0 + j];
      float bn0 = g1[kg] * (x0 - m1[kg]) * is1[kg] + b1[kg];
      float bn1 = g1[kg+1] * (x1 - m1[kg+1]) * is1[kg+1] + b1[kg+1];
      bf2 t; t.x = (__bf16)fmaxf(bn0, 0.f); t.y = (__bf16)fmaxf(bn1, 0.f);
      Bs[j][r2] = t;
    }
    __syncthreads();
    AFrag a, b;
    const int ar = wave * 16 + (lane & 15);
    const int bc = lane & 15;
#pragma unroll
    for (int t = 0; t < 4; ++t) { a.p[t] = As[ar][4*g + t]; a.p[4 + t] = As[ar][8 + 4*g + t]; }
#pragma unroll
    for (int t = 0; t < 4; ++t) { b.p[t] = Bs[bc][4*g + t]; b.p[4 + t] = Bs[bc][8 + 4*g + t]; }
    acc = __builtin_amdgcn_wmma_f32_16x16x32_bf16(false, a.v, false, b.v, (short)0, acc, false, false);
  }
  const int n = n0 + (lane & 15);
#pragma unroll
  for (int r = 0; r < 8; ++r) {
    int c = c0 + wave * 16 + r + 8 * g;
    fp[(size_t)c * N_ + n] = acc[r] + F[(size_t)n * C_ + c] + bc2[c];
  }
}

// P[n] = clip(sigmoid(fw_w . relu(bn(fp[:,n])) + fw_b), 0.05, 0.95)
__global__ __launch_bounds__(256) void k_P(
    const float* __restrict__ fp, const float* __restrict__ m2, const float* __restrict__ is2,
    const float* __restrict__ fg, const float* __restrict__ fb,
    const float* __restrict__ fww, const float* __restrict__ fwb,
    float* __restrict__ p, float* __restrict__ out)
{
  const int n = blockIdx.x * 256 + threadIdx.x;
  float acc = fwb[0];
  for (int c = 0; c < C_; ++c) {
    float x = fp[(size_t)c * N_ + n];
    float bn = fg[c] * (x - m2[c]) * is2[c] + fb[c];
    acc += fww[c] * fmaxf(bn, 0.f);
  }
  float P = 1.f / (1.f + __expf(-acc));
  P = fminf(fmaxf(P, 0.05f), 0.95f);
  p[n] = P;
  out[3 * N_ + n] = P;
}

// ---------------------------------------------------------------------------
// A[i][j] = sum_n U[n][i]*p[n]*U[n][j] + lam*sigma2*evals[i]*(i==j)
// exact fp32 WMMA 16x16x4. grid(8,2) block(128): i0=bx*16, j0=by*64
// ---------------------------------------------------------------------------
__global__ __launch_bounds__(128) void k_Abuild(
    const float* __restrict__ U, const float* __restrict__ p,
    const float* __restrict__ evals, const float* __restrict__ sumF2, float* __restrict__ A)
{
  const int i0 = blockIdx.x * 16;
  const int j0 = blockIdx.y * 64;
  __shared__ float At[16][66];  // [i_local][k]  (= U[n][i]*p[n])
  __shared__ float Bt[64][66];  // [j_local][k]
  const int tid = threadIdx.x, lane = tid & 31, wave = tid >> 5, g = lane >> 4;
  v8f acc = {0.f,0.f,0.f,0.f,0.f,0.f,0.f,0.f};

  for (int n0 = 0; n0 < N_; n0 += 64) {
    __syncthreads();
#pragma unroll
    for (int e = 0; e < 8; ++e) {
      int idx = tid + e * 128;
      int r = idx >> 4, iL = idx & 15;
      At[iL][r] = U[(size_t)(n0 + r) * K_ + i0 + iL] * p[n0 + r];
    }
#pragma unroll
    for (int e = 0; e < 32; ++e) {
      int idx = tid + e * 128;
      int r = idx >> 6, jL = idx & 63;
      Bt[jL][r] = U[(size_t)(n0 + r) * K_ + j0 + jL];
    }
    __syncthreads();
    const int ar = lane & 15;
    const int bc = wave * 16 + (lane & 15);
#pragma unroll
    for (int kk = 0; kk < 64; kk += 4) {
      v2f a, b;
      a.x = At[ar][kk + 2*g]; a.y = At[ar][kk + 2*g + 1];
      b.x = Bt[bc][kk + 2*g]; b.y = Bt[bc][kk + 2*g + 1];
      acc = __builtin_amdgcn_wmma_f32_16x16x4_f32(false, a, false, b, (short)0, acc, false, false);
    }
  }
  const float lam = 10.f * (sumF2[0] / (float)((size_t)N_ * C_));
  const int j = j0 + wave * 16 + (lane & 15);
#pragma unroll
  for (int r = 0; r < 8; ++r) {
    int i = i0 + r + 8 * g;
    float v = acc[r];
    if (i == j) v += lam * evals[i];
    A[i * 132 + j] = v;
  }
}

// rhs[i][j] = sum_n U[n][i]*p[n]*vertF[n][j], stored in augmented cols 128..130
__global__ __launch_bounds__(256) void k_rhs(
    const float* __restrict__ U, const float* __restrict__ p,
    const float* __restrict__ vertF, float* __restrict__ A)
{
  const int i = blockIdx.x, tid = threadIdx.x;
  float s0 = 0.f, s1 = 0.f, s2 = 0.f;
  for (int n = tid; n < N_; n += 256) {
    float up = U[(size_t)n * K_ + i] * p[n];
    s0 += up * vertF[n*3+0]; s1 += up * vertF[n*3+1]; s2 += up * vertF[n*3+2];
  }
  __shared__ float red[3][256];
  red[0][tid] = s0; red[1][tid] = s1; red[2][tid] = s2; __syncthreads();
  for (int o = 128; o; o >>= 1) {
    if (tid < o) { red[0][tid]+=red[0][tid+o]; red[1][tid]+=red[1][tid+o]; red[2][tid]+=red[2][tid+o]; }
    __syncthreads();
  }
  if (tid < 3) A[i * 132 + 128 + tid] = red[tid][0];
}

// Gaussian elimination (SPD, no pivoting) on augmented [A | rhs], then back-sub.
__global__ __launch_bounds__(512) void k_solve(float* __restrict__ A, float* __restrict__ W) {
  const int AP = 132;
  const int tid = threadIdx.x;
  for (int r = 0; r < K_; ++r) {
    __syncthreads();
    float piv = A[r*AP + r];
    int row = r + 1 + tid;
    if (row < K_) {
      float f = A[row*AP + r] / piv;
      for (int c = r + 1; c < K_ + 3; ++c) A[row*AP + c] -= f * A[r*AP + c];
    }
  }
  __shared__ float red[3][128];
  __shared__ float Wr[K_ * 3];
  const int j = tid >> 7, k = tid & 127;
  for (int r = K_ - 1; r >= 0; --r) {
    __syncthreads();
    float prod = 0.f;
    if (j < 3 && k > r) prod = A[r*AP + k] * Wr[k*3 + j];
    if (j < 3) red[j][k] = prod;
    __syncthreads();
    for (int s = 64; s; s >>= 1) { if (j < 3 && k < s) red[j][k] += red[j][k + s]; __syncthreads(); }
    if (j < 3 && k == 0) Wr[r*3 + j] = (A[r*AP + K_ + j] - red[j][0]) / A[r*AP + r];
  }
  __syncthreads();
  if (tid < K_ * 3) W[tid] = Wr[tid];
}

// T[n][j] = U[n][:] . W[:][j] + vert_x[n][j]
__global__ __launch_bounds__(256) void k_T(
    const float* __restrict__ U, const float* __restrict__ W,
    const float* __restrict__ vert_x, float* __restrict__ out)
{
  __shared__ float Ws[K_ * 3];
  for (int i = threadIdx.x; i < K_ * 3; i += 256) Ws[i] = W[i];
  __syncthreads();
  const int n = blockIdx.x * 256 + threadIdx.x;
  float s0 = vert_x[n*3+0], s1 = vert_x[n*3+1], s2 = vert_x[n*3+2];
  for (int i = 0; i < K_; ++i) {
    float uv = U[(size_t)n * K_ + i];
    s0 += uv * Ws[i*3+0]; s1 += uv * Ws[i*3+1]; s2 += uv * Ws[i*3+2];
  }
  out[n*3+0] = s0; out[n*3+1] = s1; out[n*3+2] = s2;
}

extern "C" void kernel_launch(void* const* d_in, const int* in_sizes, int n_in,
                              void* d_out, int out_size, void* d_ws, size_t ws_size,
                              hipStream_t stream)
{
  (void)in_sizes; (void)n_in; (void)out_size; (void)ws_size;
  const float* feat_x = (const float*)d_in[0];
  const float* feat_y = (const float*)d_in[1];
  const float* Pxy    = (const float*)d_in[2];
  const float* vert_x = (const float*)d_in[3];
  const float* vert_y = (const float*)d_in[4];
  const float* evecs  = (const float*)d_in[5];
  const float* evals  = (const float*)d_in[6];
  const float* loc    = (const float*)d_in[7];
  const float* w_up   = (const float*)d_in[8];
  const float* b_up   = (const float*)d_in[9];
  const float* wq     = (const float*)d_in[10];
  const float* bq     = (const float*)d_in[11];
  const float* wk     = (const float*)d_in[12];
  const float* wv     = (const float*)d_in[14];
  const float* bv     = (const float*)d_in[15];
  const float* wmh    = (const float*)d_in[16];
  const float* bmh    = (const float*)d_in[17];
  const float* wc1    = (const float*)d_in[18];
  const float* bc1    = (const float*)d_in[19];
  const float* bn1g   = (const float*)d_in[20];
  const float* bn1b   = (const float*)d_in[21];
  const float* wc2    = (const float*)d_in[22];
  const float* bc2    = (const float*)d_in[23];
  const float* fwg    = (const float*)d_in[24];
  const float* fwb    = (const float*)d_in[25];
  const float* fww    = (const float*)d_in[26];
  const float* fwb0   = (const float*)d_in[27];
  float* out = (float*)d_out;

  float* w = (float*)d_ws;
  float* F     = w; w += (size_t)N_ * C_;
  float* h1    = w; w += (size_t)C2_ * N_;
  float* fp    = w; w += (size_t)C_ * N_;
  float* vertF = w; w += N_ * 3;
  float* alpha = w; w += 4 * N_;
  float* Es    = w; w += 4 * N_;
  float* pbuf  = w; w += N_;
  float* u     = w; w += 4 * C_;
  float* tvec  = w; w += 4;
  float* G1    = w; w += 4 * C2_;
  float* e1    = w; w += C2_;
  float* m1    = w; w += C2_;
  float* is1   = w; w += C2_;
  float* m2    = w; w += C_;
  float* is2   = w; w += C_;
  float* Amat  = w; w += 128 * 132;
  float* Wsol  = w; w += 384;
  float* part  = w; w += 512;
  float* sumF2 = w; w += 1;

  k_gemmF<<<dim3(4, 96), 256, 0, stream>>>(Pxy, feat_y, feat_x, F, part);
  k_sumfin<<<1, 128, 0, stream>>>(part, sumF2);
  k_vertF<<<N_, 256, 0, stream>>>(Pxy, vert_y, vert_x, vertF);
  k_pre<<<1, 512, 0, stream>>>(wq, bq, wk, wv, bv, wmh, bmh, wc1, bc1, w_up, b_up,
                               u, tvec, G1, e1);
  k_alpha<<<N_, 256, 0, stream>>>(F, u, tvec, alpha);
  k_es<<<3072, 256, 0, stream>>>(alpha, loc, Es);
  k_h1<<<dim3(4, 384), 256, 0, stream>>>(wc1, F, Es, G1, e1, h1);
  k_stats<<<C2_, 256, 0, stream>>>(h1, m1, is1);
  k_fp<<<dim3(2, 384), 256, 0, stream>>>(wc2, h1, m1, is1, bn1g, bn1b, F, bc2, fp);
  k_stats<<<C_, 256, 0, stream>>>(fp, m2, is2);
  k_P<<<24, 256, 0, stream>>>(fp, m2, is2, fwg, fwb, fww, fwb0, pbuf, out);
  k_Abuild<<<dim3(8, 2), 128, 0, stream>>>(evecs, pbuf, evals, sumF2, Amat);
  k_rhs<<<128, 256, 0, stream>>>(evecs, pbuf, vertF, Amat);
  k_solve<<<1, 512, 0, stream>>>(Amat, Wsol);
  k_T<<<24, 256, 0, stream>>>(evecs, Wsol, vert_x, out);
}
